// DiagonalLSTM_42399917146486
// MI455X (gfx1250) — compile-verified, matrix operand order checked by reading, not run on hardware
//
#include <hip/hip_runtime.h>
#include <hip/hip_bf16.h>
#include <stdint.h>

// ---------------------------------------------------------------------------
// DiagonalLSTM on MI455X (gfx1250, wave32, WMMA)
//   gates(640x64) = Wcat(640x320) @ [x_diag(64x64); h_sh(128x64); h(128x64)]
//   via v_wmma_f32_16x16x32_bf16, fp32 accumulation, fp32 cell state.
// ---------------------------------------------------------------------------

#define IN_CH   64
#define HID     128
#define HH      64
#define WW      64
#define BB      16
#define TT      127          // H + W - 1 diagonals
#define M5      640          // 5 * HID gate rows
#define KTOT    320          // 64 (x) + 128 (h shifted) + 128 (h)
#define HSTR    136          // hbuf row stride in bf16 (272 B, 16B aligned)
#define CSTR    66           // cbuf row stride in floats

typedef __attribute__((ext_vector_type(16))) __bf16   v16bf;
typedef __attribute__((ext_vector_type(8)))  float    v8f;
typedef __attribute__((ext_vector_type(4)))  uint32_t u32x4;

// global (addrspace 1) pointer to a 16-byte chunk — keeps laundered loads
// on the GLOBAL path (LOADcnt only), not FLAT (LOADcnt+DScnt).
typedef const __attribute__((address_space(1))) u32x4* gp4;

union FragB { v16bf v; u32x4 q[2]; };
union FragF { v8f   v; u32x4 q[2]; };

__device__ __forceinline__ v8f wmma_bf16(v16bf a, v16bf b, v8f c) {
  // (neg_a, A, neg_b, B, c_mod, C, reuse_a, reuse_b)
  return __builtin_amdgcn_wmma_f32_16x16x32_bf16(false, a, false, b,
                                                 (short)0, c, false, false);
}

__device__ __forceinline__ float fsig(float x) {
  return 1.f / (1.f + __expf(-x));
}
__device__ __forceinline__ float ftanh(float x) {
  float e = __expf(-2.f * x);
  return (1.f - e) / (1.f + e);
}

// ---------------------------------------------------------------------------
// Prep: pack [W_is | W0 | W1] as bf16 row-major 640x320, fused bias.
// ---------------------------------------------------------------------------
__global__ void prep_weights(const float* __restrict__ w_is,
                             const float* __restrict__ b_is,
                             const float* __restrict__ w_ss,
                             const float* __restrict__ b_ss,
                             __bf16* __restrict__ Wcat,
                             float* __restrict__ biascat) {
  int idx = blockIdx.x * 256 + threadIdx.x;
  if (idx < M5) biascat[idx] = b_is[idx] + b_ss[idx];
  if (idx >= M5 * KTOT) return;
  int m = idx / KTOT, k = idx % KTOT;
  float v;
  if (k < 64)        v = w_is[m * 64 + k];                    // input->gates
  else if (k < 192)  v = w_ss[(m * HID + (k - 64)) * 2 + 0];  // w0 (h shifted)
  else               v = w_ss[(m * HID + (k - 192)) * 2 + 1]; // w1 (h)
  Wcat[idx] = (__bf16)v;
}

// ---------------------------------------------------------------------------
// Prep: diag-major bf16 x: xskew[b][t][r][c] = x[b][c][r][t-r] (0 if invalid)
// ---------------------------------------------------------------------------
__global__ void prep_xskew(const float* __restrict__ x,
                           __bf16* __restrict__ xskew) {
  size_t idx = (size_t)blockIdx.x * 256 + threadIdx.x;
  if (idx >= (size_t)BB * TT * HH * 64) return;
  int c = (int)(idx & 63);
  size_t tmp = idx >> 6;
  int r = (int)(tmp & 63); tmp >>= 6;
  int t = (int)(tmp % TT);
  int b = (int)(tmp / TT);
  int w = t - r;
  float v = (w >= 0 && w < WW)
                ? x[(((size_t)b * IN_CH + c) * HH + r) * WW + w] : 0.f;
  xskew[idx] = (__bf16)v;
}

// ---------------------------------------------------------------------------
// Residual: out = w_res @ x + b_res  (fully rewrites d_out every launch)
// ---------------------------------------------------------------------------
__global__ void residual_kernel(const float* __restrict__ x,
                                const float* __restrict__ w_res,
                                const float* __restrict__ b_res,
                                float* __restrict__ out) {
  int b = blockIdx.x / HH, r = blockIdx.x % HH;
  __shared__ float xs[64][65];
  for (int i = threadIdx.x; i < 64 * 64; i += 256) {
    int c = i >> 6, w = i & 63;
    xs[c][w] = x[(((size_t)b * IN_CH + c) * HH + r) * WW + w];
  }
  __syncthreads();
  for (int oo = threadIdx.x; oo < HID * WW; oo += 256) {
    int o = oo >> 6, w = oo & 63;
    float acc = b_res[o];
#pragma unroll
    for (int c = 0; c < 64; ++c) acc += w_res[o * 64 + c] * xs[c][w];
    out[(((size_t)b * HID + o) * HH + r) * WW + w] = acc;
  }
}

// ---------------------------------------------------------------------------
// Recurrence: one workgroup per batch, 16 waves (512 threads).
//   wave = (m0, ngrp): m0 = channel block (16 ch), ngrp = H-column half.
//   Acc tiles [n][G]: G in {o, f_left, f_up, i, g} for the SAME 16 channels,
//   so the LSTM cell update is lane-local across the five gate tiles.
// ---------------------------------------------------------------------------
__global__ __launch_bounds__(512, 1) void dlstm_kernel(
    const __bf16* __restrict__ xskew, const __bf16* __restrict__ Wcat,
    const float* __restrict__ biascat, float* __restrict__ out) {
  __shared__ __align__(16) __bf16 hbuf[HH * HSTR];   // h state, bf16 [r][j]
  __shared__ float cbuf[HID * CSTR];                 // c state, f32 [j][r]
  __shared__ __align__(16) float biasl[M5];          // fused bias (LDS copy)

  const int b    = blockIdx.x;
  const int tid  = threadIdx.x;
  const int wave = tid >> 5;
  const int lane = tid & 31;
  const int lo   = lane & 15;
  const int hi   = lane >> 4;
  const int m0   = wave >> 1;   // 0..7  channel block
  const int ngrp = wave & 1;    // 0..1  column half

  for (int i = tid; i < HH * HSTR; i += 512) hbuf[i] = (__bf16)0.f;
  for (int i = tid; i < HID * CSTR; i += 512) cbuf[i] = 0.f;
  if (tid < M5) biasl[tid] = biascat[tid];
  if (tid + 512 < M5) biasl[tid + 512] = biascat[tid + 512];
  __syncthreads();

  const int r0 = 32 * ngrp + lo;       // N row of tile n=0 (n=1 is r0+16)
  const int jlane = 16 * m0 + 8 * hi;  // channel base for C/D fragments

  // Flat base pointers; all fragment loads below use immediate offsets.
  const uint64_t waddr =
      (uint64_t)(Wcat + (size_t)(16 * m0 + lo) * KTOT + 8 * hi);
  const float*  blds = &biasl[16 * m0 + 8 * hi];          // +128*G
  const __bf16* hb0  = &hbuf[r0 * HSTR + 8 * hi];         // h row r0
  const __bf16* hb0m = &hbuf[(r0 - 1) * HSTR + 8 * hi];   // h row r0-1 (r0>0)
  const __bf16* hb1  = &hbuf[(r0 + 16) * HSTR + 8 * hi];
  const __bf16* hb1m = &hbuf[(r0 + 15) * HSTR + 8 * hi];

  const u32x4 zero4 = {0u, 0u, 0u, 0u};

  for (int t = 0; t < TT; ++t) {
    const __bf16* xp = xskew + ((((size_t)b * TT + t) * HH + r0) * 64) + 8 * hi;
    if (t + 1 < TT) __builtin_prefetch(xp + HH * 64, 0, 3);  // next diagonal

    // Launder the weight base so the L2-resident weight stream is re-issued
    // every step instead of being hoisted (and spilled) by LICM. Cast back
    // to addrspace(1) so the loads stay on the GLOBAL path.
    uint64_t wa = waddr;
    asm volatile("" : "+v"(wa));

    // ---- acc init = fused bias (LDS; barrier-clobbered, not hoistable) -
    v8f acc[2][5];
#pragma unroll
    for (int G = 0; G < 5; ++G) {
      FragF f;
      f.q[0] = *(const u32x4*)(blds + 128 * G);
      f.q[1] = *(const u32x4*)(blds + 128 * G + 4);
      acc[0][G] = f.v;
      acc[1][G] = f.v;
    }

    // ---- GEMM: 10 k-tiles of K=32 -------------------------------------
#pragma unroll
    for (int kt = 0; kt < 10; ++kt) {
      FragB B0, B1;
      if (kt < 2) {                       // x diagonal column (global/L2)
        B0.q[0] = *(const u32x4*)(xp + 32 * kt);
        B0.q[1] = *(const u32x4*)(xp + 32 * kt + 16);
        B1.q[0] = *(const u32x4*)(xp + 1024 + 32 * kt);        // r0+16 row
        B1.q[1] = *(const u32x4*)(xp + 1024 + 32 * kt + 16);
      } else if (kt < 6) {                // h shifted by one H row (LDS)
        const int ko = 32 * (kt - 2);
        if (r0 == 0) { B0.q[0] = zero4; B0.q[1] = zero4; }
        else {
          B0.q[0] = *(const u32x4*)(hb0m + ko);
          B0.q[1] = *(const u32x4*)(hb0m + ko + 16);
        }
        B1.q[0] = *(const u32x4*)(hb1m + ko);
        B1.q[1] = *(const u32x4*)(hb1m + ko + 16);
      } else {                            // h (LDS)
        const int ko = 32 * (kt - 6);
        B0.q[0] = *(const u32x4*)(hb0 + ko);
        B0.q[1] = *(const u32x4*)(hb0 + ko + 16);
        B1.q[0] = *(const u32x4*)(hb1 + ko);
        B1.q[1] = *(const u32x4*)(hb1 + ko + 16);
      }
#pragma unroll
      for (int G = 0; G < 5; ++G) {       // A: one gate fragment at a time
        FragB A;
        A.q[0] = *(gp4)(wa + (size_t)G * 81920 + 64 * kt);        // bytes
        A.q[1] = *(gp4)(wa + (size_t)G * 81920 + 64 * kt + 32);
        acc[0][G] = wmma_bf16(A.v, B0.v, acc[0][G]);
        acc[1][G] = wmma_bf16(A.v, B1.v, acc[1][G]);
      }
    }

    __syncthreads();  // all GEMM reads of hbuf done

    // ---- stage c_prev / c_up before anyone overwrites cbuf ------------
    float cp[2][8], cl[2][8];
#pragma unroll
    for (int n = 0; n < 2; ++n) {
      const int r = r0 + 16 * n;
#pragma unroll
      for (int v = 0; v < 8; ++v) {
        const int j = jlane + v;
        cp[n][v] = cbuf[j * CSTR + r];
        cl[n][v] = (r > 0) ? cbuf[j * CSTR + r - 1] : 0.f;
      }
    }
    __syncthreads();  // c reads done

    // ---- LSTM cell update; write new h/c; emit valid outputs ----------
#pragma unroll
    for (int n = 0; n < 2; ++n) {
      const int r = r0 + 16 * n;
      const int wcol = t - r;
#pragma unroll
      for (int v = 0; v < 8; ++v) {
        const int j = jlane + v;
        float o  = fsig(acc[n][0][v]);
        float fl = fsig(acc[n][1][v]);
        float fu = fsig(acc[n][2][v]);
        float ii = fsig(acc[n][3][v]);
        float g  = ftanh(acc[n][4][v]);
        float c  = fl * cp[n][v] + fu * cl[n][v] + ii * g;
        float h  = o * ftanh(c);
        cbuf[j * CSTR + r] = c;
        hbuf[r * HSTR + j] = (__bf16)h;
        if (wcol >= 0 && wcol < WW)
          out[(((size_t)b * HID + j) * HH + r) * WW + wcol] += h;
      }
    }
    __syncthreads();  // state visible before next step's GEMM
  }
}

// ---------------------------------------------------------------------------
extern "C" void kernel_launch(void* const* d_in, const int* in_sizes, int n_in,
                              void* d_out, int out_size, void* d_ws,
                              size_t ws_size, hipStream_t stream) {
  const float* x     = (const float*)d_in[0];
  const float* w_is  = (const float*)d_in[1];
  const float* b_is  = (const float*)d_in[2];
  const float* w_ss  = (const float*)d_in[3];
  const float* b_ss  = (const float*)d_in[4];
  const float* w_res = (const float*)d_in[5];
  const float* b_res = (const float*)d_in[6];
  float* out = (float*)d_out;

  char* ws = (char*)d_ws;
  __bf16* Wcat    = (__bf16*)ws;                 // 640*320*2   = 409,600 B
  float*  biascat = (float*)(ws + 409600);       // 640*4       =   2,560 B
  __bf16* xskew   = (__bf16*)(ws + 425984);      // 16*127*64*64*2 = 16.6 MB

  prep_weights<<<(M5 * KTOT + 255) / 256, 256, 0, stream>>>(
      w_is, b_is, w_ss, b_ss, Wcat, biascat);

  size_t nsk = (size_t)BB * TT * HH * 64;
  prep_xskew<<<(int)((nsk + 255) / 256), 256, 0, stream>>>(x, xskew);

  residual_kernel<<<BB * HH, 256, 0, stream>>>(x, w_res, b_res, out);

  dlstm_kernel<<<BB, 512, 0, stream>>>(xskew, Wcat, biascat, out);
}
